// GRUD_69922067578967
// MI455X (gfx1250) — matrix operand order repaired and checked by reading, not compiled, not run on hardware
//
#include <hip/hip_runtime.h>

// GRU-D on MI455X (gfx1250): bf16 WMMA for the three H x H recurrent GEMMs,
// f32 accumulate + f32 pointwise. Weights/state are L2-resident (~5 MB);
// only the 256 MB of output states stream to HBM (~11 us at 23.3 TB/s).

#define I_DIM 512
#define H_DIM 512
#define T_DIM 256

typedef __attribute__((ext_vector_type(16))) __bf16 v16bf;
typedef __attribute__((ext_vector_type(8)))  float  v8f;

__device__ __forceinline__ unsigned short f2bf(float f) {
    unsigned int u = __float_as_uint(f);
    unsigned int r = u + 0x7fffu + ((u >> 16) & 1u);   // round-to-nearest-even
    return (unsigned short)(r >> 16);
}

// A fragment: 16-bit A matrix 16x32 (MxK), ISA 7.12.2 layout.
// lane l holds row m = l&15; half h: K = 2*(p&3) + 16*(p>>2) + 8*(l>>4) (+k0), pairs contiguous.
__device__ __forceinline__ v16bf load_fragA(const unsigned short* __restrict__ A,
                                            int row, int k0, int lane) {
    union { unsigned int u[8]; v16bf v; } f;
    const unsigned int* p = (const unsigned int*)(A + (size_t)row * H_DIM);
    const int hi = (lane >> 4) & 1;
#pragma unroll
    for (int pi = 0; pi < 8; ++pi) {
        int k = k0 + 2 * (pi & 3) + ((pi >> 2) << 4) + (hi << 3);
        f.u[pi] = p[k >> 1];
    }
    return f.v;
}

// B fragment: 16-bit B matrix 32x16 (KxN). lane l holds col n = l&15;
// half h: K = 2*(p&3) + 16*(p>>2) + 8*(l>>4) (+k0), matching fragment A.
// BT is stored transposed (N x K row-major) so K pairs are contiguous.
__device__ __forceinline__ v16bf load_fragB(const unsigned short* __restrict__ BT,
                                            int col, int k0, int lane) {
    union { unsigned int u[8]; v16bf v; } f;
    const unsigned int* p = (const unsigned int*)(BT + (size_t)col * H_DIM);
    const int hi = (lane >> 4) & 1;
#pragma unroll
    for (int pi = 0; pi < 8; ++pi) {
        int k = k0 + 2 * (pi & 3) + ((pi >> 2) << 4) + (hi << 3);
        f.u[pi] = p[k >> 1];
    }
    return f.v;
}

__device__ __forceinline__ float sigmoidf(float x) { return 1.0f / (1.0f + expf(-x)); }

// ---- one-time: W^T bf16 conversion of w_hr, w_hz, w_hh -------------------
__global__ void __launch_bounds__(256)
grud_convert(const float* __restrict__ w_hr, const float* __restrict__ w_hz,
             const float* __restrict__ w_hh,
             unsigned short* __restrict__ wTr, unsigned short* __restrict__ wTz,
             unsigned short* __restrict__ wTh) {
    int idx = blockIdx.x * 256 + threadIdx.x;          // 0 .. 3*H*H-1
    int sel = idx / (H_DIM * H_DIM);
    int e   = idx - sel * (H_DIM * H_DIM);
    int n = e / H_DIM;
    int k = e - n * H_DIM;
    const float* src = (sel == 0) ? w_hr : (sel == 1) ? w_hz : w_hh;
    unsigned short* dst = (sel == 0) ? wTr : (sel == 1) ? wTz : wTh;
    dst[(size_t)n * H_DIM + k] = f2bf(src[(size_t)k * H_DIM + n]);
}

// ---- per step, pass 1: decay h, imputed x ---------------------------------
// block = row i (512 blocks x 256 threads, 2 cols/thread)
__global__ void __launch_bounds__(256)
grud_pass1(const float* __restrict__ data, int t,
           const float* __restrict__ h_prev,
           const float* __restrict__ w_dg_x, const float* __restrict__ w_dg_h,
           const float* __restrict__ b_dg_x, const float* __restrict__ b_dg_h,
           float* __restrict__ hdec, unsigned int* __restrict__ hdec_bf,
           float* __restrict__ xs, float* __restrict__ ms) {
    const int i   = blockIdx.x;
    const int tid = threadIdx.x;
    const size_t TI = (size_t)T_DIM * I_DIM;
    const float xv = data[0 * TI + (size_t)t * I_DIM + i];
    const float mt = data[1 * TI + (size_t)t * I_DIM + i];
    const float dd = data[2 * TI + (size_t)t * I_DIM + i];
    const float xp = data[3 * TI + (size_t)t * I_DIM + i];
    if (tid == 0) {
        float gx = expf(-fmaxf(0.0f, w_dg_x[i] * dd + b_dg_x[i]));
        xs[i] = mt * xv + (1.0f - mt) * (gx * xp);
        ms[i] = mt;
    }
    const int k0 = 2 * tid;
    const size_t base = (size_t)i * H_DIM + k0;
    float g0 = expf(-fmaxf(0.0f, dd * w_dg_h[k0]     + b_dg_h[base]));
    float g1 = expf(-fmaxf(0.0f, dd * w_dg_h[k0 + 1] + b_dg_h[base + 1]));
    float h0 = g0 * h_prev[base];
    float h1 = g1 * h_prev[base + 1];
    hdec[base]     = h0;
    hdec[base + 1] = h1;
    hdec_bf[base >> 1] = (unsigned int)f2bf(h0) | ((unsigned int)f2bf(h1) << 16);
}

// ---- per step, pass 2: fused r/z GEMM (dual WMMA, shared A) ---------------
// 1024 tiles of 16x16, one per wave; 8 waves/block -> 128 blocks.
__global__ void __launch_bounds__(256)
grud_pass2(const unsigned short* __restrict__ hdec_bf,
           const unsigned short* __restrict__ wTr, const unsigned short* __restrict__ wTz,
           const float* __restrict__ xs, const float* __restrict__ ms,
           const float* __restrict__ w_xr, const float* __restrict__ w_mr,
           const float* __restrict__ w_xz, const float* __restrict__ w_mz,
           const float* __restrict__ b_r, const float* __restrict__ b_z,
           const float* __restrict__ hdec,
           unsigned short* __restrict__ rh_bf, float* __restrict__ zbuf) {
    const int lane = threadIdx.x & 31;
    const int tile = blockIdx.x * 8 + (threadIdx.x >> 5);
    const int tm = tile >> 5, tn = tile & 31;
    v8f accr = {}; v8f accz = {};
    const int arow = tm * 16 + (lane & 15);
    const int bcol = tn * 16 + (lane & 15);
#pragma unroll 4
    for (int k0 = 0; k0 < H_DIM; k0 += 32) {
        v16bf a  = load_fragA(hdec_bf, arow, k0, lane);
        v16bf br = load_fragB(wTr, bcol, k0, lane);
        v16bf bz = load_fragB(wTz, bcol, k0, lane);
        accr = __builtin_amdgcn_wmma_f32_16x16x32_bf16(false, a, false, br,
                                                       (short)0, accr, false, false);
        accz = __builtin_amdgcn_wmma_f32_16x16x32_bf16(false, a, false, bz,
                                                       (short)0, accz, false, false);
    }
    const int col = tn * 16 + (lane & 15);
    const int hi  = lane >> 4;
    const float wxr = w_xr[col], wmr = w_mr[col];
    const float wxz = w_xz[col], wmz = w_mz[col];
#pragma unroll
    for (int v = 0; v < 8; ++v) {
        const int row = tm * 16 + hi * 8 + v;        // C/D layout: M = v + 8*(lane>=16)
        const size_t e = (size_t)row * H_DIM + col;
        float r = sigmoidf(accr[v] + xs[row] * wxr + ms[row] * wmr + b_r[e]);
        float z = sigmoidf(accz[v] + xs[row] * wxz + ms[row] * wmz + b_z[e]);
        zbuf[e]  = z;
        rh_bf[e] = f2bf(r * hdec[e]);
    }
}

// ---- per step, pass 3: h_tilde GEMM + blend -------------------------------
__global__ void __launch_bounds__(256)
grud_pass3(const unsigned short* __restrict__ rh_bf,
           const unsigned short* __restrict__ wTh,
           const float* __restrict__ xs, const float* __restrict__ ms,
           const float* __restrict__ w_xh, const float* __restrict__ w_mh,
           const float* __restrict__ b_h,
           const float* __restrict__ zbuf, const float* __restrict__ hdec,
           float* __restrict__ out_t) {
    const int lane = threadIdx.x & 31;
    const int tile = blockIdx.x * 8 + (threadIdx.x >> 5);
    const int tm = tile >> 5, tn = tile & 31;
    v8f acc = {};
    const int arow = tm * 16 + (lane & 15);
    const int bcol = tn * 16 + (lane & 15);
#pragma unroll 4
    for (int k0 = 0; k0 < H_DIM; k0 += 32) {
        v16bf a = load_fragA(rh_bf, arow, k0, lane);
        v16bf b = load_fragB(wTh, bcol, k0, lane);
        acc = __builtin_amdgcn_wmma_f32_16x16x32_bf16(false, a, false, b,
                                                      (short)0, acc, false, false);
    }
    const int col = tn * 16 + (lane & 15);
    const int hi  = lane >> 4;
    const float wxh = w_xh[col], wmh = w_mh[col];
#pragma unroll
    for (int v = 0; v < 8; ++v) {
        const int row = tm * 16 + hi * 8 + v;
        const size_t e = (size_t)row * H_DIM + col;
        float ht = tanhf(acc[v] + xs[row] * wxh + ms[row] * wmh + b_h[e]);
        float z  = zbuf[e];
        out_t[e] = (1.0f - z) * hdec[e] + z * ht;
    }
}

extern "C" void kernel_launch(void* const* d_in, const int* in_sizes, int n_in,
                              void* d_out, int out_size, void* d_ws, size_t ws_size,
                              hipStream_t stream) {
    const float* data   = (const float*)d_in[0];
    const float* state  = (const float*)d_in[1];
    const float* w_dg_x = (const float*)d_in[2];
    const float* w_dg_h = (const float*)d_in[3];
    const float* w_xr   = (const float*)d_in[4];
    const float* w_hr   = (const float*)d_in[5];
    const float* w_mr   = (const float*)d_in[6];
    const float* w_xz   = (const float*)d_in[7];
    const float* w_hz   = (const float*)d_in[8];
    const float* w_mz   = (const float*)d_in[9];
    const float* w_xh   = (const float*)d_in[10];
    const float* w_hh   = (const float*)d_in[11];
    const float* w_mh   = (const float*)d_in[12];
    const float* b_dg_x = (const float*)d_in[13];
    const float* b_dg_h = (const float*)d_in[14];
    const float* b_r    = (const float*)d_in[15];
    const float* b_z    = (const float*)d_in[16];
    const float* b_h    = (const float*)d_in[17];
    float* out = (float*)d_out;

    char* ws = (char*)d_ws;
    const size_t MB = 1u << 20;
    float*          hdec    = (float*)(ws);                         // 1 MB
    float*          zbuf    = (float*)(ws + 1 * MB);                // 1 MB
    unsigned short* hdec_bf = (unsigned short*)(ws + 2 * MB);       // 0.5 MB
    unsigned short* rh_bf   = (unsigned short*)(ws + 2 * MB + 512 * 1024); // 0.5 MB
    unsigned short* wTr     = (unsigned short*)(ws + 3 * MB);       // 0.5 MB
    unsigned short* wTz     = (unsigned short*)(ws + 3 * MB + 512 * 1024);
    unsigned short* wTh     = (unsigned short*)(ws + 4 * MB);
    float*          xs      = (float*)(ws + 4 * MB + 512 * 1024);
    float*          ms      = xs + I_DIM;

    grud_convert<<<(3 * H_DIM * H_DIM) / 256, 256, 0, stream>>>(
        w_hr, w_hz, w_hh, wTr, wTz, wTh);

    for (int t = 0; t < T_DIM; ++t) {
        const float* h_prev = (t == 0) ? state : out + (size_t)(t - 1) * I_DIM * H_DIM;
        grud_pass1<<<I_DIM, 256, 0, stream>>>(data, t, h_prev,
                                              w_dg_x, w_dg_h, b_dg_x, b_dg_h,
                                              hdec, (unsigned int*)hdec_bf, xs, ms);
        grud_pass2<<<128, 256, 0, stream>>>(hdec_bf, wTr, wTz, xs, ms,
                                            w_xr, w_mr, w_xz, w_mz, b_r, b_z,
                                            hdec, rh_bf, zbuf);
        grud_pass3<<<128, 256, 0, stream>>>(rh_bf, wTh, xs, ms,
                                            w_xh, w_mh, b_h, zbuf, hdec,
                                            out + (size_t)t * I_DIM * H_DIM);
    }
}
